// SelfAttention_53558242181618
// MI455X (gfx1250) — compile-verified, matrix operand order checked by reading, not run on hardware
//
#include <hip/hip_runtime.h>
#include <hip/hip_bf16.h>

typedef __bf16 bf16_t;
typedef __attribute__((ext_vector_type(16))) __bf16 bf16x16;
typedef __attribute__((ext_vector_type(8)))  __bf16 bf16x8;
typedef __attribute__((ext_vector_type(8)))  float  f32x8;
typedef __attribute__((ext_vector_type(4)))  unsigned int u32x4;
typedef __attribute__((ext_vector_type(8)))  int  i32x8;
typedef __attribute__((ext_vector_type(4)))  int  i32x4;

#define B_   4
#define S_   2048
#define H_   1024
#define NH_  16
#define HD_  64

#define GLD  72   // GEMM LDS leading dim (64 + 8 pad), 144B rows, 16B aligned
#define KLD  72   // attn K-tile leading dim (64 + 8 pad)
#define VLD  40   // attn V-tile leading dim (32 + 8 pad), 80B rows
#define PLD  40   // attn P staging leading dim

// Q pre-scale: HD^-0.5 * log2(e); scores leave WMMA already in log2 domain.
#define QSCL 0.1803368787f

// ---------------------------------------------------------------------------
// Fragment loader for 16x32 bf16 tiles, row-major source with leading dim ld.
// Lane L holds row (row0 + L%16); elements 0..7 hold K = k0 + (L>=16?8:0)+0..7,
// elements 8..15 hold K = +16 of that. Two contiguous 16B loads per lane.
// ---------------------------------------------------------------------------
__device__ __forceinline__ bf16x16 load_frag(const bf16_t* base,
                                             int ld, int row0, int k0) {
    const int lane = threadIdx.x & 31;
    const int row  = row0 + (lane & 15);
    const int kk   = k0 + ((lane >> 4) << 3);
    const bf16_t* p = base + (long)row * ld + kk;
    union { bf16x16 v; bf16x8 h[2]; } u;
    u.h[0] = *(const bf16x8*)(p);
    u.h[1] = *(const bf16x8*)(p + 16);
    return u.v;
}

__device__ __forceinline__ f32x8 wmma_bf16(bf16x16 a, bf16x16 b, f32x8 c) {
    return __builtin_amdgcn_wmma_f32_16x16x32_bf16(false, a, false, b,
                                                   (short)0, c, false, false);
}

// Async global->LDS 16B copy (GLOBAL_LOAD_ASYNC_TO_LDS_B128, ASYNCcnt).
__device__ __forceinline__ void async_cp16(bf16_t* lds_dst, const bf16_t* gsrc) {
    const unsigned int       d = (unsigned int)(size_t)lds_dst;   // LDS offset
    const unsigned long long s = (unsigned long long)(size_t)gsrc;
    asm volatile("global_load_async_to_lds_b128 %0, %1, off"
                 :: "v"(d), "v"(s) : "memory");
}
__device__ __forceinline__ void wait_async0() {
    asm volatile("s_wait_asynccnt 0x0" ::: "memory");
}

// ---------------------------------------------------------------------------
// TDM: DMA a 2-D bf16 tile (rows x 64) from a row-major matrix (leading dim
// ldK elements) into LDS, inserting 16B of padding after every 128B row so the
// LDS image has leading dim GLD=72 (bank-conflict-free fragment reads).
// D# layout per CDNA5 ISA 8.3/8.4; 6-arg builtin (g0,g1,g2,g3,g4,cpol).
// ---------------------------------------------------------------------------
__device__ __forceinline__ void tdm_load_2d(const bf16_t* gsrc, bf16_t* lds_dst,
                                            int rows, int ldK) {
    const unsigned long long ga = (unsigned long long)(size_t)gsrc;
    u32x4 g0;
    g0[0] = 1u;                                        // count=1, user, load
    g0[1] = (unsigned int)(size_t)lds_dst;             // lds_addr
    g0[2] = (unsigned int)(ga & 0xFFFFFFFFull);        // global_addr[31:0]
    g0[3] = (unsigned int)((ga >> 32) & 0x01FFFFFFu)   // global_addr[56:32]
          | (2u << 30);                                // type = 2 (image)
    i32x8 g1;
    g1[0] = (1 << 16)                                  // data_size = 2 bytes
          | (1 << 20)                                  // pad_enable
          | (4 << 22)                                  // pad_interval: 128B
          | (3 << 25);                                 // pad_amount: 4 DW = 16B
    g1[1] = (int)(((unsigned)ldK & 0xFFFFu) << 16);    // tensor_dim0 lo16
    g1[2] = (int)((0xFFFFu << 16) | (((unsigned)ldK >> 16) & 0xFFFFu));
                                                       // tensor_dim1 lo16 = max
    g1[3] = (64 << 16);                                // tile_dim0 = 64
    g1[4] = rows & 0xFFFF;                             // tile_dim1; tile_dim2=0
    g1[5] = ldK;                                       // tensor_dim0_stride lo
    g1[6] = 0;
    g1[7] = 0;
    const i32x4 z4 = {0, 0, 0, 0};
    const i32x8 z8 = {0, 0, 0, 0, 0, 0, 0, 0};
    __builtin_amdgcn_tensor_load_to_lds(g0, g1, z4, z4, z8, 0);
}
__device__ __forceinline__ void wait_tensor0() {
    __builtin_amdgcn_s_wait_tensorcnt(0);
}

// DPP ROW_XMASK lane-xor within groups of 16 (wave32): pure-VALU reductions.
template <int CTRL>
__device__ __forceinline__ float dppmov(float x) {
    return __int_as_float(__builtin_amdgcn_update_dpp(
        0, __float_as_int(x), CTRL, 0xf, 0xf, true));
}
__device__ __forceinline__ float red_max16(float x) {
    x = fmaxf(x, dppmov<0x161>(x));
    x = fmaxf(x, dppmov<0x162>(x));
    x = fmaxf(x, dppmov<0x164>(x));
    x = fmaxf(x, dppmov<0x168>(x));
    return x;
}
__device__ __forceinline__ float red_sum16(float x) {
    x += dppmov<0x161>(x);
    x += dppmov<0x162>(x);
    x += dppmov<0x164>(x);
    x += dppmov<0x168>(x);
    return x;
}

// ---------------------------------------------------------------------------
// fp32 -> bf16 convert / transpose helpers
// ---------------------------------------------------------------------------
__global__ void k_cvt_bf16(const float* __restrict__ in,
                           bf16_t* __restrict__ out, int n) {
    int i = blockIdx.x * blockDim.x + threadIdx.x;
    if (i < n) out[i] = (bf16_t)in[i];
}

// in: [K,N] fp32 row-major -> out: [N,K] bf16 row-major
__global__ void k_transpose_bf16(const float* __restrict__ in,
                                 bf16_t* __restrict__ out, int K, int N) {
    int idx = blockIdx.x * blockDim.x + threadIdx.x;
    if (idx >= K * N) return;
    int n = idx / K;
    int k = idx - n * K;
    out[idx] = (bf16_t)in[k * N + n];
}

// ---------------------------------------------------------------------------
// LDS-tiled GEMM: C[M,N] = A[M,K] * BT[N,K]^T (+bias).
// Block tile 128(M) x 128(N), 8 waves as 4x2, wave tile 32x64.
// Double-buffered TDM staging of 64-wide k-slabs (each wave DMAs its own
// 16-row stripe of A and B, waits on its own TENSORcnt). 16 WMMA per stage.
// MODE 0: scatter into Q/K/V^T (Q pre-scaled).  MODE 1: fp32 out + bias.
// ---------------------------------------------------------------------------
template <int MODE>
__global__ __launch_bounds__(256) void k_gemm(
    const bf16_t* __restrict__ A,    // [M, K]
    const bf16_t* __restrict__ BT,   // [N, K]
    const float*  __restrict__ bias, // [N]
    float*  __restrict__ outf,       // MODE 1: [M, N]
    bf16_t* __restrict__ qm,         // MODE 0: [B*NH, S, HD]
    bf16_t* __restrict__ km,
    bf16_t* __restrict__ vtm,        // [B*NH, HD, S]
    int M, int N, int K)
{
    __shared__ alignas(16) bf16_t As[2][128 * GLD];
    __shared__ alignas(16) bf16_t Bs[2][128 * GLD];

    const int wave = threadIdx.x >> 5;
    const int lane = threadIdx.x & 31;
    const int M0 = blockIdx.y * 128;
    const int N0 = blockIdx.x * 128;
    const int mw = (wave >> 1) * 32;   // wave row offset in block tile
    const int nw = (wave & 1) * 64;    // wave col offset in block tile

    // per-wave TDM fetch of this wave's 16-row stripe of a 128x64 k-slab
    auto tile_fetch = [&](const bf16_t* g, int row0, int k0, bf16_t* lds) {
        tdm_load_2d(g + (long)(row0 + wave * 16) * K + k0,
                    lds + wave * 16 * GLD, 16, K);
    };

    f32x8 acc[2][4] = {};

    tile_fetch(A,  M0, 0, As[0]);
    tile_fetch(BT, N0, 0, Bs[0]);
    wait_tensor0();
    __syncthreads();

    int cur = 0;
    for (int k0 = 0; k0 < K; k0 += 64) {
        const int nxt = cur ^ 1;
        if (k0 + 64 < K) {
            tile_fetch(A,  M0, k0 + 64, As[nxt]);
            tile_fetch(BT, N0, k0 + 64, Bs[nxt]);
        }
#pragma unroll
        for (int kk = 0; kk < 64; kk += 32) {
            bf16x16 af[2], bfr[4];
#pragma unroll
            for (int i = 0; i < 2; ++i)
                af[i] = load_frag(As[cur], GLD, mw + 16 * i, kk);
#pragma unroll
            for (int j = 0; j < 4; ++j)
                bfr[j] = load_frag(Bs[cur], GLD, nw + 16 * j, kk);
#pragma unroll
            for (int i = 0; i < 2; ++i)
#pragma unroll
                for (int j = 0; j < 4; ++j)
                    acc[i][j] = wmma_bf16(af[i], bfr[j], acc[i][j]);
        }
        wait_tensor0();
        __syncthreads();
        cur = nxt;
    }

    const int colb   = lane & 15;
    const int rowoff = (lane >> 4) * 8;
#pragma unroll
    for (int i = 0; i < 2; ++i) {
#pragma unroll
        for (int j = 0; j < 4; ++j) {
            const int n    = N0 + nw + 16 * j + colb;
            const float bv = bias[n];
            if (MODE == 1) {
#pragma unroll
                for (int r = 0; r < 8; ++r) {
                    const long m = M0 + mw + 16 * i + rowoff + r;
                    outf[m * N + n] = acc[i][j][r] + bv;
                }
            } else {
                const int which = n >> 10;        // 0=q 1=k 2=v
                const int hh    = n & 1023;
                const int head  = hh >> 6;
                const int d     = hh & 63;
#pragma unroll
                for (int r = 0; r < 8; ++r) {
                    const int m  = M0 + mw + 16 * i + rowoff + r;
                    const int bb = m >> 11;       // / S
                    const int s  = m & 2047;
                    const long bh = (long)bb * NH_ + head;
                    float val = acc[i][j][r] + bv;
                    if (which == 0) {
                        val *= QSCL;              // fold softmax scale into Q
                        qm[bh * (S_ * HD_) + (long)s * HD_ + d] = (bf16_t)val;
                    } else if (which == 1) {
                        km[bh * (S_ * HD_) + (long)s * HD_ + d] = (bf16_t)val;
                    } else {
                        vtm[bh * (HD_ * S_) + (long)d * S_ + s] = (bf16_t)val;
                    }
                }
            }
        }
    }
}

// ---------------------------------------------------------------------------
// Flash attention: block owns (b,h) x 128 q-rows; 8 waves x 16 q-rows each.
// K/V^T tiles double-buffered in LDS via async copies, shared by all waves.
// Q pre-scaled -> scores already in log2 domain; DPP row reductions; interior
// tiles skip the causal mask entirely.
// ---------------------------------------------------------------------------
__global__ __launch_bounds__(256) void k_attn(
    const bf16_t* __restrict__ qm,   // [BH, S, HD] (pre-scaled)
    const bf16_t* __restrict__ km,   // [BH, S, HD]
    const bf16_t* __restrict__ vtm,  // [BH, HD, S]
    bf16_t* __restrict__ out)        // [B*S, H]
{
    __shared__ alignas(16) bf16_t Ks[2][32 * KLD];    // rows = k, cols = d
    __shared__ alignas(16) bf16_t Vs[2][64 * VLD];    // rows = d, cols = s
    __shared__ alignas(16) bf16_t Pbuf[8][16 * PLD];  // per-wave P staging

    const int wave = threadIdx.x >> 5;
    const int lane = threadIdx.x & 31;
    const int qblk = blockIdx.x & 15;    // S/128 = 16 q-blocks
    const int bh   = blockIdx.x >> 4;    // 0..63
    const int bb   = bh >> 4;
    const int h    = bh & 15;
    const int q0   = qblk * 128 + wave * 16;

    const bf16_t* qp = qm  + (long)bh * (S_ * HD_);
    const bf16_t* kp = km  + (long)bh * (S_ * HD_);
    const bf16_t* vp = vtm + (long)bh * (HD_ * S_);
    bf16_t* P = Pbuf[wave];

    // cooperative K/V tile fetch for k-range [kt, kt+32)
    auto tile_fetch = [&](int kt, int buf) {
        const int t = threadIdx.x;
        {   // K tile: 32 rows x 64 d (source contiguous 4KB)
            const int row = t >> 3;
            const int dc  = (t & 7) * 8;
            async_cp16(Ks[buf] + row * KLD + dc,
                       kp + (long)(kt + row) * HD_ + dc);
        }
        {   // V^T tile: 64 rows (d) x 32 cols (s), row stride S_
            const int row = t >> 2;
            const int sc  = (t & 3) * 8;
            async_cp16(Vs[buf] + row * VLD + sc,
                       vp + (long)row * S_ + kt + sc);
        }
    };

    const bf16x16 aq0 = load_frag(qp, HD_, q0, 0);
    const bf16x16 aq1 = load_frag(qp, HD_, q0, 32);

    f32x8 accO[4] = {};
    float mrow[8], lrow[8];
#pragma unroll
    for (int r = 0; r < 8; ++r) { mrow[r] = -__builtin_inff(); lrow[r] = 0.f; }

    const int colb   = lane & 15;
    const int rowoff = (lane >> 4) * 8;
    const int KTend  = qblk * 128 + 128;        // block-uniform k bound

    tile_fetch(0, 0);
    wait_async0();
    __syncthreads();

    int cur = 0;
    for (int kt = 0; kt < KTend; kt += 32) {
        const int nxt = cur ^ 1;
        if (kt + 32 < KTend) tile_fetch(kt + 32, nxt);

        if (kt < q0 + 16) {   // causal: this wave still has live columns
            // ---- scores (log2 domain): two 16x16 tiles ------------------
            f32x8 sc[2];
#pragma unroll
            for (int jj = 0; jj < 2; ++jj) {
                f32x8 cacc = {};
                bf16x16 bk0 = load_frag(Ks[cur], KLD, 16 * jj, 0);
                cacc = wmma_bf16(aq0, bk0, cacc);
                bf16x16 bk1 = load_frag(Ks[cur], KLD, 16 * jj, 32);
                cacc = wmma_bf16(aq1, bk1, cacc);
                sc[jj] = cacc;
            }
            // ---- (mask only on diagonal tiles) + row max ----------------
            float tmax[8];
            if (kt + 31 <= q0) {       // interior tile: no masking needed
#pragma unroll
                for (int r = 0; r < 8; ++r)
                    tmax[r] = fmaxf(sc[0][r], sc[1][r]);
            } else {                   // diagonal tile: per-element mask
#pragma unroll
                for (int r = 0; r < 8; ++r) {
                    const int qrow = q0 + rowoff + r;
                    float s0 = (kt +      colb <= qrow) ? sc[0][r]
                                                        : -__builtin_inff();
                    float s1 = (kt + 16 + colb <= qrow) ? sc[1][r]
                                                        : -__builtin_inff();
                    sc[0][r] = s0; sc[1][r] = s1;
                    tmax[r] = fmaxf(s0, s1);
                }
            }
#pragma unroll
            for (int r = 0; r < 8; ++r) tmax[r] = red_max16(tmax[r]);

            // ---- online softmax update (all in log2 domain) -------------
            float alpha[8];
#pragma unroll
            for (int r = 0; r < 8; ++r) {
                const float mnew = fmaxf(mrow[r], tmax[r]);
                alpha[r] = exp2f(mrow[r] - mnew);
                mrow[r]  = mnew;
            }
#pragma unroll
            for (int r = 0; r < 8; ++r) {
                const float p0 = exp2f(sc[0][r] - mrow[r]);
                const float p1 = exp2f(sc[1][r] - mrow[r]);
                const float rs = red_sum16(p0 + p1);
                lrow[r] = lrow[r] * alpha[r] + rs;
                const int prow = rowoff + r;
                P[prow * PLD + colb]      = (bf16_t)p0;   // C-layout -> LDS
                P[prow * PLD + colb + 16] = (bf16_t)p1;
            }
            // same-wave DS store -> DS fragment load ordering
            asm volatile("s_wait_dscnt 0x0" ::: "memory");

            // ---- PV: P[16x32] * V^T-tile[32x64] -------------------------
            const bf16x16 pa = load_frag(P, PLD, 0, 0);
#pragma unroll
            for (int j = 0; j < 4; ++j) {
#pragma unroll
                for (int r = 0; r < 8; ++r) accO[j][r] *= alpha[r];
                bf16x16 bv = load_frag(Vs[cur], VLD, 16 * j, 0);
                accO[j] = wmma_bf16(pa, bv, accO[j]);
            }
        }

        wait_async0();
        __syncthreads();
        cur = nxt;
    }

    // ---- finalize: divide by l, store bf16 [B*S, H] ----------------------
#pragma unroll
    for (int r = 0; r < 8; ++r) lrow[r] = 1.0f / lrow[r];
    const long mbase = (long)bb * S_ + q0;
#pragma unroll
    for (int j = 0; j < 4; ++j) {
        const int c = h * HD_ + 16 * j + colb;
#pragma unroll
        for (int r = 0; r < 8; ++r) {
            const long m = mbase + rowoff + r;
            out[m * H_ + c] = (bf16_t)(accO[j][r] * lrow[r]);
        }
    }
}

// ---------------------------------------------------------------------------
extern "C" void kernel_launch(void* const* d_in, const int* in_sizes, int n_in,
                              void* d_out, int out_size, void* d_ws, size_t ws_size,
                              hipStream_t stream) {
    const float* x     = (const float*)d_in[0];   // [B,S,H]
    const float* w_qkv = (const float*)d_in[1];   // [H,3H]
    const float* b_qkv = (const float*)d_in[2];   // [3H]
    const float* w_out = (const float*)d_in[3];   // [H,H]
    const float* b_out = (const float*)d_in[4];   // [H]
    float* out = (float*)d_out;                   // [B,S,H] fp32

    const int M  = B_ * S_;   // 8192
    const int K  = H_;        // 1024
    const int N1 = 3 * H_;    // 3072

    size_t off = 0;
    auto carve = [&](size_t bytes) {
        void* p = (char*)d_ws + off;
        off += (bytes + 255) & ~(size_t)255;
        return p;
    };
    bf16_t* xb    = (bf16_t*)carve((size_t)M * K * 2);
    bf16_t* wqkvT = (bf16_t*)carve((size_t)N1 * K * 2);
    bf16_t* woutT = (bf16_t*)carve((size_t)H_ * H_ * 2);
    bf16_t* qb    = (bf16_t*)carve((size_t)B_ * NH_ * S_ * HD_ * 2);
    bf16_t* kb    = (bf16_t*)carve((size_t)B_ * NH_ * S_ * HD_ * 2);
    bf16_t* vtb   = (bf16_t*)carve((size_t)B_ * NH_ * S_ * HD_ * 2);
    bf16_t* attn  = xb;   // xb dead after QKV GEMM; reuse for attention output

    // 1) operand conversion
    {
        int n = M * K;
        k_cvt_bf16<<<(n + 255) / 256, 256, 0, stream>>>(x, xb, n);
    }
    {
        int n = K * N1;
        k_transpose_bf16<<<(n + 255) / 256, 256, 0, stream>>>(w_qkv, wqkvT, K, N1);
    }
    {
        int n = K * H_;
        k_transpose_bf16<<<(n + 255) / 256, 256, 0, stream>>>(w_out, woutT, K, H_);
    }

    // 2) QKV projection -> Q (pre-scaled), K, V^T (heads layout)
    {
        dim3 grid(N1 / 128, M / 128);
        k_gemm<0><<<grid, 256, 0, stream>>>(xb, wqkvT, b_qkv, nullptr,
                                            qb, kb, vtb, M, N1, K);
    }

    // 3) causal flash attention -> attn [M, H] bf16
    {
        k_attn<<<(B_ * NH_) * (S_ / 128), 256, 0, stream>>>(qb, kb, vtb, attn);
    }

    // 4) output projection -> fp32 d_out
    {
        dim3 grid(H_ / 128, M / 128);
        k_gemm<1><<<grid, 256, 0, stream>>>(attn, woutT, b_out, out,
                                            nullptr, nullptr, nullptr, M, H_, K);
    }
}
